// GasStream_82806969467503
// MI455X (gfx1250) — matrix operand order repaired and checked by reading, not compiled, not run on hardware
//
#include <hip/hip_runtime.h>
#include <hip/hip_bf16.h>
#include <cstdint>

// ---------------------------------------------------------------------------
// MI455X / gfx1250.  wave32, v_wmma_f32_16x16x32_bf16 (fp32 accumulate),
// async global->LDS for the feat stream, LDS-staged BN scale/shift.
// ---------------------------------------------------------------------------
typedef __attribute__((ext_vector_type(16))) __bf16 v16bf;
typedef __attribute__((ext_vector_type(2)))  __bf16 v2bf;
typedef __attribute__((ext_vector_type(8)))  float  v8f;

#define BATCH   8192
#define T_N     2
#define M_N     3
#define F_N     40
#define E_N     64
#define K2E     128          // 2*E
#define FP      48           // F padded to WMMA N multiple
#define FE      2560         // F*E
#define EPSV    1e-5f

// LDS bank padding (64 banks x 4B)
#define FS_STRIDE   2568     // k_main feat tile per-b stride (bf16 elems)
#define FS_STRIDE_F 2564     // k_stats feat tile per-b stride (f32 elems, 16B-aligned rows)
#define NY_BSTRIDE  3080     // nei/y tile per-b stride (bf16 elems)

// ---- workspace byte layout (total ~434 KB) --------------------------------
#define WS_ADJ   0                       // T*64*FP bf16           = 12288
#define WS_W     12288                   // T*K2E*E bf16           = 32768
#define WS_FGW   (12288 + 32768)         // FE*FP bf16 (K-major)   = 245760
#define WS_S1    (WS_FGW + 245760)       // F*E   f32 sum(feat)    = 10240
#define WS_Q1    (WS_S1 + 10240)         // F*E   f32 sum(feat^2)
#define WS_S2    (WS_Q1 + 10240)         // T*F*E f32 sum(nei)     = 20480
#define WS_Q2    (WS_S2 + 20480)         // T*F*E f32 sum(nei^2)
#define WS_SC    (WS_Q2 + 20480)         // T*F*K2E f32 bn scale   = 40960
#define WS_SH    (WS_SC + 40960)         // T*F*K2E f32 bn shift   = 40960

__device__ __forceinline__ v8f wmma_bf16(v16bf a, v16bf b, v8f c) {
  return __builtin_amdgcn_wmma_f32_16x16x32_bf16(false, a, false, b,
                                                 (short)0, c, false, false);
}

// ===========================================================================
// Kernel 1: adjacency pipeline (tiny) + bf16 weight packing + zero stats
// ===========================================================================
__global__ void __launch_bounds__(256)
k_prep(const float* __restrict__ masker,
       const float* __restrict__ ln_gamma, const float* __restrict__ ln_beta,
       const float* __restrict__ gnn_w,    const float* __restrict__ fgate_w,
       unsigned char* __restrict__ ws) {
  __bf16* adj_bf = (__bf16*)(ws + WS_ADJ);
  __bf16* w_bf   = (__bf16*)(ws + WS_W);
  __bf16* fgw_bf = (__bf16*)(ws + WS_FGW);
  float*  stats  = (float*)(ws + WS_S1);     // S1,Q1,S2,Q2 contiguous
  const int tid = threadIdx.x;

  for (int i = tid; i < T_N * 64 * FP; i += 256) adj_bf[i] = (__bf16)0.f;
  for (int i = tid; i < 15360;         i += 256) stats[i]  = 0.f;
  for (int i = tid; i < T_N * K2E * E_N; i += 256) w_bf[i] = (__bf16)gnn_w[i];
  // fgate weights stored K-major [k][fo(pad48)] so B-fragment lane reads are
  // contiguous across the 16 lanes of a half-wave.
  for (int i = tid; i < FE * FP; i += 256) {
    const int k = i / FP, fo = i % FP;
    fgw_bf[i] = (fo < F_N) ? (__bf16)fgate_w[fo * FE + k] : (__bf16)0.f;
  }
  __syncthreads();

  if (tid < T_N * F_N) {
    const int t = tid / F_N, g = tid % F_N;
    float col[F_N], lg[F_N];
    float mu = 0.f;
    for (int f = 0; f < F_N; ++f) {
      float p = 1.f;
      for (int m = 0; m < M_N; ++m)
        p *= masker[((t * M_N + m) * F_N + f) * F_N + g];
      p = p > 0.f ? p : 0.f;
      col[f] = p; mu += p;
    }
    mu *= (1.f / F_N);
    float var = 0.f;
    for (int f = 0; f < F_N; ++f) { float d = col[f] - mu; var += d * d; }
    var *= (1.f / F_N);
    const float rs = rsqrtf(var + EPSV);
    float mx = -1e30f;
    for (int f = 0; f < F_N; ++f) {
      const float msk = (col[f] != 0.f) ? 1.f : 0.f;
      float v = (col[f] - mu) * rs * ln_gamma[f] + ln_beta[f];
      v += (1.f - msk) * (-1e9f) + ((f == g) ? 1.f : 0.f);
      lg[f] = v; mx = fmaxf(mx, v);
    }
    float sum = 0.f;
    for (int f = 0; f < F_N; ++f) { lg[f] = __expf(lg[f] - mx); sum += lg[f]; }
    const float inv = 1.f / sum;
    for (int f = 0; f < F_N; ++f) {
      const float msk = (col[f] != 0.f) ? 1.f : 0.f;
      // WMMA B-operand layout: row-major [K=f pad64][N=g pad48]
      adj_bf[(t * 64 + f) * FP + g] = (__bf16)(lg[f] * inv * msk);
    }
  }
}

// ===========================================================================
// Kernel 2: batch statistics.  feat tile arrives via async DMA to LDS
// (ASYNCcnt path); nei via WMMA (M=batch16, N=g, K=f padded 64).
// ===========================================================================
__global__ void __launch_bounds__(128)
k_stats(const float* __restrict__ feat, unsigned char* __restrict__ ws) {
  const __bf16* adj_bf = (const __bf16*)(ws + WS_ADJ);
  float* S1 = (float*)(ws + WS_S1);
  float* Q1 = (float*)(ws + WS_Q1);
  float* S2 = (float*)(ws + WS_S2);
  float* Q2 = (float*)(ws + WS_Q2);

  __shared__ float feat_s[16 * FS_STRIDE_F];       // 164,096 B, fp32
  const int tid  = threadIdx.x;
  const int b0   = blockIdx.x * 16;
  const int lane = tid & 31, w = tid >> 5;
  const int h = lane >> 4, ln15 = lane & 15;

  // ---- async global -> LDS, 16B per lane per issue (GLOBAL_LOAD_ASYNC) ----
  for (int c = tid; c < 16 * (FE / 4); c += 128) {
    const int b = c / (FE / 4), j = c % (FE / 4);
    const uint32_t ldsoff = (uint32_t)(uintptr_t)(feat_s + b * FS_STRIDE_F + j * 4);
    const float*   src    = feat + (size_t)(b0 + b) * FE + (size_t)j * 4;
    asm volatile("global_load_async_to_lds_b128 %0, %1, off"
                 :: "v"(ldsoff), "v"(src) : "memory");
  }
  asm volatile("s_wait_asynccnt 0" ::: "memory");
  __syncthreads();

  // first-half channels: stats of feat per (f,e)  (fp32 accuracy)
  for (int c = tid; c < FE; c += 128) {
    float s = 0.f, q = 0.f;
    for (int b = 0; b < 16; ++b) {
      const float v = feat_s[b * FS_STRIDE_F + c];
      s += v; q += v * v;
    }
    atomicAdd(&S1[c], s); atomicAdd(&Q1[c], q);
  }

  // second-half channels: nei = feat[b,:,e] @ adj[t]
  for (int t = 0; t < T_N; ++t) {
    v16bf adjB[3][2];
    for (int gt = 0; gt < 3; ++gt)
      for (int kc = 0; kc < 2; ++kc) {
#pragma unroll
        for (int i = 0; i < 16; ++i) {
          const int k = kc * 32 + ((i >> 3) << 4) + h * 8 + (i & 7); // B frag K map
          adjB[gt][kc][i] = adj_bf[(t * 64 + k) * FP + gt * 16 + ln15];
        }
      }
    for (int ei = 0; ei < 16; ++ei) {
      const int e = w + 4 * ei;                               // wave e-slice
      v16bf a0, a1;
#pragma unroll
      for (int i = 0; i < 16; ++i) {
        const int k0 = ((i >> 3) << 4) + h * 8 + (i & 7);     // A frag K map
        const int k1 = 32 + k0;
        a0[i] = (k0 < F_N) ? (__bf16)feat_s[ln15 * FS_STRIDE_F + k0 * E_N + e]
                           : (__bf16)0.f;
        a1[i] = (k1 < F_N) ? (__bf16)feat_s[ln15 * FS_STRIDE_F + k1 * E_N + e]
                           : (__bf16)0.f;
      }
      for (int gt = 0; gt < 3; ++gt) {
        v8f d = {};
        d = wmma_bf16(a0, adjB[gt][0], d);
        d = wmma_bf16(a1, adjB[gt][1], d);
        const int g = gt * 16 + ln15;
        if (g < F_N) {
          float s = 0.f, q = 0.f;
#pragma unroll
          for (int j = 0; j < 8; ++j) { s += d[j]; q += d[j] * d[j]; }
          atomicAdd(&S2[(t * F_N + g) * E_N + e], s);
          atomicAdd(&Q2[(t * F_N + g) * E_N + e], q);
        }
      }
    }
  }
}

// ===========================================================================
// Kernel 3: fold batch stats + bn gamma/beta into per-channel scale/shift
// ===========================================================================
__global__ void __launch_bounds__(256)
k_final(const float* __restrict__ bn_gamma, const float* __restrict__ bn_beta,
        unsigned char* __restrict__ ws) {
  const float* S1 = (const float*)(ws + WS_S1);
  const float* Q1 = (const float*)(ws + WS_Q1);
  const float* S2 = (const float*)(ws + WS_S2);
  const float* Q2 = (const float*)(ws + WS_Q2);
  float* SC = (float*)(ws + WS_SC);
  float* SH = (float*)(ws + WS_SH);
  const float invB = 1.f / (float)BATCH;
  for (int c = threadIdx.x; c < T_N * F_N * K2E; c += 256) {
    const int r = c % (F_N * K2E);
    const int t = c / (F_N * K2E), f = r / K2E, k = r % K2E;
    float m, q;
    if (k < E_N) { m = S1[f * E_N + k] * invB; q = Q1[f * E_N + k] * invB; }
    else {
      const int idx = (t * F_N + f) * E_N + (k - E_N);
      m = S2[idx] * invB; q = Q2[idx] * invB;
    }
    const float sc = bn_gamma[c] * rsqrtf(q - m * m + EPSV);
    SC[c] = sc; SH[c] = bn_beta[c] - m * sc;
  }
}

// ===========================================================================
// Kernel 4: fused nei -> BN -> GNN GEMM -> fgate GEMM/softmax -> tgate -> out
// One workgroup (4 waves) per 16 batch rows.  ~229 KB LDS of 320 KB/WGP.
// ===========================================================================
__global__ void __launch_bounds__(128)
k_main(const float* __restrict__ feat,
       const float* __restrict__ gnn_b,  const float* __restrict__ fgate_b,
       const float* __restrict__ tgate_w, const float* __restrict__ tgate_b,
       const unsigned char* __restrict__ ws, float* __restrict__ out) {
  const __bf16* adj_bf = (const __bf16*)(ws + WS_ADJ);
  const __bf16* w_bf   = (const __bf16*)(ws + WS_W);
  const __bf16* fgw_bf = (const __bf16*)(ws + WS_FGW);
  const float*  SC     = (const float*)(ws + WS_SC);
  const float*  SH     = (const float*)(ws + WS_SH);

  __shared__ __bf16 feat_s[16 * FS_STRIDE];   // 82,176 B : feat tile (bf16)
  __shared__ __bf16 ny_s[16 * NY_BSTRIDE];    // 98,560 B : nei(norm) then y
  __shared__ float  sc_s[F_N * K2E];          // 20,480 B : BN scale (cur t)
  __shared__ float  sh_s[F_N * K2E];          // 20,480 B : BN shift (cur t)
  __shared__ float  lg_s[16 * FP];            // fgate logits
  __shared__ float  tw_s[16 * T_N];           // tower gate weights
  __shared__ float  acc_s[16 * E_N];          // output accumulator
  __shared__ float  red_s[128];

  const int tid  = threadIdx.x;
  const int b0   = blockIdx.x * 16;
  const int lane = tid & 31, w = tid >> 5;
  const int h = lane >> 4, ln15 = lane & 15;

  for (int i = tid; i < 16 * FE; i += 128) {
    const int b = i / FE, j = i % FE;
    feat_s[b * FS_STRIDE + j] = (__bf16)feat[(size_t)(b0 + b) * FE + j];
  }
  for (int i = tid; i < 16 * E_N; i += 128) acc_s[i] = 0.f;
  __syncthreads();

  // ---- tower gate: softmax_t( feat_flat . tgate_w[t] + b ) ----
  {
    const int p = tid >> 2, q4 = tid & 3;       // p -> (b,t)
    const int b = p >> 1, t = p & 1;
    float s = 0.f;
    for (int k = q4; k < FE; k += 4)
      s += (float)feat_s[b * FS_STRIDE + k] * tgate_w[t * FE + k];
    red_s[tid] = s;
  }
  __syncthreads();
  if (tid < 32) {
    const int b = tid >> 1, t = tid & 1;
    tw_s[b * 2 + t] = red_s[tid * 4] + red_s[tid * 4 + 1] +
                      red_s[tid * 4 + 2] + red_s[tid * 4 + 3] + tgate_b[t];
  }
  __syncthreads();
  if (tid < 16) {
    const float l0 = tw_s[tid * 2], l1 = tw_s[tid * 2 + 1];
    const float mx = fmaxf(l0, l1);
    const float e0 = __expf(l0 - mx), e1 = __expf(l1 - mx);
    const float inv = 1.f / (e0 + e1);
    tw_s[tid * 2] = e0 * inv; tw_s[tid * 2 + 1] = e1 * inv;
  }
  __syncthreads();

  for (int t = 0; t < T_N; ++t) {
    // ---- stage BN scale/shift for this tower into LDS (coalesced) --------
    for (int i = tid; i < F_N * K2E; i += 128) {
      sc_s[i] = SC[t * F_N * K2E + i];
      sh_s[i] = SH[t * F_N * K2E + i];
    }
    __syncthreads();

    // ---------------- nei (WMMA) + BatchNorm -> ny_s (bf16) ----------------
    v16bf adjB[3][2];
    for (int gt = 0; gt < 3; ++gt)
      for (int kc = 0; kc < 2; ++kc) {
#pragma unroll
        for (int i = 0; i < 16; ++i) {
          const int k = kc * 32 + ((i >> 3) << 4) + h * 8 + (i & 7);
          adjB[gt][kc][i] = adj_bf[(t * 64 + k) * FP + gt * 16 + ln15];
        }
      }
    for (int ei = 0; ei < 16; ++ei) {
      const int e = w + 4 * ei;
      v16bf a0, a1;
#pragma unroll
      for (int i = 0; i < 16; ++i) {
        const int k0 = ((i >> 3) << 4) + h * 8 + (i & 7);
        const int k1 = 32 + k0;
        a0[i] = (k0 < F_N) ? feat_s[ln15 * FS_STRIDE + k0 * E_N + e] : (__bf16)0.f;
        a1[i] = (k1 < F_N) ? feat_s[ln15 * FS_STRIDE + k1 * E_N + e] : (__bf16)0.f;
      }
      for (int gt = 0; gt < 3; ++gt) {
        v8f d = {};
        d = wmma_bf16(a0, adjB[gt][0], d);
        d = wmma_bf16(a1, adjB[gt][1], d);
        const int g = gt * 16 + ln15;
        if (g < F_N) {
          const float sc = sc_s[g * K2E + E_N + e];
          const float sh = sh_s[g * K2E + E_N + e];
#pragma unroll
          for (int j = 0; j < 8; ++j) {
            const int b = j + 8 * h;
            ny_s[b * NY_BSTRIDE + g * E_N + e] = (__bf16)(d[j] * sc + sh);
          }
        }
      }
    }
    __syncthreads();

    // ---------------- GNN GEMM: y[b,f,:] = xc_norm[b,f,:] @ W[t] + bias ----
    v16bf wB[4];
    {
      const int eN = w * 16 + ln15;          // wave owns e-tile w
      for (int kc = 0; kc < 4; ++kc) {
#pragma unroll
        for (int i = 0; i < 16; ++i) {
          const int k = kc * 32 + ((i >> 3) << 4) + h * 8 + (i & 7);
          wB[kc][i] = w_bf[(t * K2E + k) * E_N + eN];
        }
      }
    }
    for (int f = 0; f < F_N; ++f) {
      const v2bf* ftrow = (const v2bf*)(feat_s + ln15 * FS_STRIDE + f * E_N);
      const v2bf* nyrow = (const v2bf*)(ny_s  + ln15 * NY_BSTRIDE + f * E_N);
      v16bf a[4];
#pragma unroll
      for (int kc = 0; kc < 2; ++kc) {        // feat half: normalize on the fly
#pragma unroll
        for (int ii = 0; ii < 8; ++ii) {
          const int i = ii * 2;
          const int k = kc * 32 + ((i >> 3) << 4) + h * 8 + (i & 7);  // even
          const v2bf pr = ftrow[k >> 1];
          a[kc][i]     = (__bf16)((float)pr.x * sc_s[f * K2E + k]     + sh_s[f * K2E + k]);
          a[kc][i + 1] = (__bf16)((float)pr.y * sc_s[f * K2E + k + 1] + sh_s[f * K2E + k + 1]);
        }
      }
#pragma unroll
      for (int kc = 2; kc < 4; ++kc) {        // nei half: already normalized
#pragma unroll
        for (int ii = 0; ii < 8; ++ii) {
          const int i = ii * 2;
          const int k = kc * 32 + ((i >> 3) << 4) + h * 8 + (i & 7);  // even
          const v2bf pr = nyrow[(k - E_N) >> 1];
          a[kc][i] = pr.x; a[kc][i + 1] = pr.y;
        }
      }
      v8f d = {};
      d = wmma_bf16(a[0], wB[0], d);
      d = wmma_bf16(a[1], wB[1], d);
      d = wmma_bf16(a[2], wB[2], d);
      d = wmma_bf16(a[3], wB[3], d);
      __syncthreads();   // all waves done reading nei row f before overwrite
      {
        const int e = w * 16 + ln15;
        const float bias = gnn_b[(t * F_N + f) * E_N + e];
#pragma unroll
        for (int j = 0; j < 8; ++j) {
          const int b = j + 8 * h;
          ny_s[b * NY_BSTRIDE + f * E_N + e] = (__bf16)(d[j] + bias); // y
        }
      }
    }

    // ---------------- field gate: logits = Y_flat @ fgw^T  (WMMA) ----------
    for (int i = tid; i < 16 * FP; i += 128) lg_s[i] = 0.f;
    __syncthreads();
    {
      const v2bf* nyb = (const v2bf*)(ny_s + ln15 * NY_BSTRIDE);
      v8f dl[3] = {v8f{}, v8f{}, v8f{}};
      for (int kcw = 0; kcw < 20; ++kcw) {   // waves split K: 80 chunks of 32
        const int kc = w * 20 + kcw;
        if (kcw + 1 < 20)                    // global_prefetch_b8 next chunk
          __builtin_prefetch(&fgw_bf[(kc + 1) * 32 * FP], 0, 1);
        v16bf a;
#pragma unroll
        for (int ii = 0; ii < 8; ++ii) {     // (k>>6)*64 + (k&63) == k
          const int i = ii * 2;
          const int k = kc * 32 + ((i >> 3) << 4) + h * 8 + (i & 7);  // even
          const v2bf pr = nyb[k >> 1];
          a[i] = pr.x; a[i + 1] = pr.y;
        }
        for (int nt = 0; nt < 3; ++nt) {
          v16bf bfr;
#pragma unroll
          for (int i = 0; i < 16; ++i) {
            const int k = kc * 32 + ((i >> 3) << 4) + h * 8 + (i & 7);
            bfr[i] = fgw_bf[k * FP + nt * 16 + ln15];   // K-major: coalesced
          }
          dl[nt] = wmma_bf16(a, bfr, dl[nt]);
        }
      }
      for (int nt = 0; nt < 3; ++nt) {
        const int fo = nt * 16 + ln15;
        if (fo < F_N) {
#pragma unroll
          for (int j = 0; j < 8; ++j)
            atomicAdd(&lg_s[(j + 8 * h) * FP + fo], dl[nt][j]);
        }
      }
    }
    __syncthreads();

    if (tid < 16) {                          // softmax over fields per b
      const int b = tid;
      float mx = -1e30f;
      for (int f = 0; f < F_N; ++f) mx = fmaxf(mx, lg_s[b * FP + f] + fgate_b[f]);
      float sum = 0.f;
      for (int f = 0; f < F_N; ++f) {
        const float ev = __expf(lg_s[b * FP + f] + fgate_b[f] - mx);
        lg_s[b * FP + f] = ev; sum += ev;
      }
      const float inv = 1.f / sum;
      for (int f = 0; f < F_N; ++f) lg_s[b * FP + f] *= inv;
    }
    __syncthreads();

    {                                        // out += tw[t] * sum_f fw*y
      const int b = tid >> 3;
      const int e0 = (tid & 7) * 8;
      float acc[8] = {0, 0, 0, 0, 0, 0, 0, 0};
      for (int f = 0; f < F_N; ++f) {
        const float wv = lg_s[b * FP + f];
#pragma unroll
        for (int e = 0; e < 8; ++e)
          acc[e] += wv * (float)ny_s[b * NY_BSTRIDE + f * E_N + e0 + e];
      }
      const float tw = tw_s[b * 2 + t];
#pragma unroll
      for (int e = 0; e < 8; ++e) acc_s[b * E_N + e0 + e] += tw * acc[e];
    }
    __syncthreads();
  } // t

  for (int i = tid; i < 16 * E_N; i += 128)
    out[(size_t)(b0 + (i >> 6)) * E_N + (i & 63)] = acc_s[i];
}

// ===========================================================================
extern "C" void kernel_launch(void* const* d_in, const int* in_sizes, int n_in,
                              void* d_out, int out_size, void* d_ws, size_t ws_size,
                              hipStream_t stream) {
  const float* feat     = (const float*)d_in[0];
  const float* masker   = (const float*)d_in[1];
  const float* gnn_w    = (const float*)d_in[2];
  const float* gnn_b    = (const float*)d_in[3];
  const float* ln_gamma = (const float*)d_in[4];
  const float* ln_beta  = (const float*)d_in[5];
  const float* bn_gamma = (const float*)d_in[6];
  const float* bn_beta  = (const float*)d_in[7];
  const float* fgate_w  = (const float*)d_in[8];
  const float* fgate_b  = (const float*)d_in[9];
  const float* tgate_w  = (const float*)d_in[10];
  const float* tgate_b  = (const float*)d_in[11];
  unsigned char* ws = (unsigned char*)d_ws;
  float* outp = (float*)d_out;

  k_prep <<<1, 256, 0, stream>>>(masker, ln_gamma, ln_beta, gnn_w, fgate_w, ws);
  k_stats<<<BATCH / 16, 128, 0, stream>>>(feat, ws);
  k_final<<<1, 256, 0, stream>>>(bn_gamma, bn_beta, ws);
  k_main <<<BATCH / 16, 128, 0, stream>>>(feat, gnn_b, fgate_b, tgate_w,
                                          tgate_b, ws, outp);
}
